// GraphTransformerLayer_5592047419422
// MI455X (gfx1250) — matrix operand order repaired and checked by reading, not compiled, not run on hardware
//
#include <hip/hip_runtime.h>
#include <math.h>

typedef __attribute__((ext_vector_type(16))) _Float16 v16h;
typedef __attribute__((ext_vector_type(8)))  _Float16 v8h;
typedef __attribute__((ext_vector_type(8)))  float    v8f;

// ---------------------------------------------------------------------------
// WMMA helpers (CDNA5 v_wmma_f32_16x16x32_f16, wave32)
// ---------------------------------------------------------------------------
__device__ __forceinline__ v8f wmma16(v16h a, v16h b, v8f c) {
  // 8 args: (neg_a, A, neg_b, B, c_mod, C, reuse_a, reuse_b)
  return __builtin_amdgcn_wmma_f32_16x16x32_f16(false, a, false, b, (short)0, c,
                                                false, false);
}

// A fragment: 16x32 f16, source row-major in LDS (row stride = lda halves).
// lane group g = lane>>4, m = lane&15.
// half j -> K = k0 + 16g + j
__device__ __forceinline__ v16h load_a_lds(const _Float16* __restrict__ src,
                                           int m, int k0, int lda, int grp) {
  v16h a;
  const _Float16* p = src + m * lda + k0 + 16 * grp;
#pragma unroll
  for (int j = 0; j < 8; ++j) a[j] = p[j];
#pragma unroll
  for (int j = 8; j < 16; ++j) a[j] = p[j];
  return a;
}

// B fragment: 32x16, W is fp32 row-major [K][cout]; lanes 0-15: K=k0..k0+15,
// lanes 16-31: K=k0+16..k0+31; N = n (per lane).
__device__ __forceinline__ v16h load_b_gl(const float* __restrict__ W,
                                          int k0, int nn, int cout, int grp) {
  v16h b;
  const float* p = W + (size_t)(k0 + 16 * grp) * cout + nn;
#pragma unroll
  for (int j = 0; j < 16; ++j) b[j] = (_Float16)p[(size_t)j * cout];
  return b;
}

// monotonic float<->uint encoding for atomicMax on floats (incl. negatives)
__device__ __forceinline__ unsigned fenc(float f) {
  unsigned u = __float_as_uint(f);
  return ((int)u < 0) ? ~u : (u | 0x80000000u);
}
__device__ __forceinline__ float fdec(unsigned u) {
  unsigned b = (u & 0x80000000u) ? (u & 0x7fffffffu) : ~u;
  return __uint_as_float(b);
}

// ---------------------------------------------------------------------------
// Kernel 1: q,k,v = x @ {Wq,Wk,Wv}, stored f16 (q pre-scaled by 1/sqrt(16))
// block = 128 thr (4 waves), one 16-row tile; each wave does 2 col-tiles.
// ---------------------------------------------------------------------------
__device__ __forceinline__ void qkv_one(const _Float16* __restrict__ xa,
                                        const float* __restrict__ W,
                                        _Float16* __restrict__ O, int row0,
                                        int wave, int grp, int ln, float scale,
                                        int n) {
#pragma unroll
  for (int nt = 0; nt < 2; ++nt) {
    int n0 = (wave * 2 + nt) * 16;
    v8f acc = {};
#pragma unroll
    for (int ks = 0; ks < 4; ++ks) {
      v16h a = load_a_lds(xa, ln, ks * 32, 128, grp);
      v16h b = load_b_gl(W, ks * 32, n0 + ln, 128, grp);
      acc = wmma16(a, b, acc);
    }
#pragma unroll
    for (int r = 0; r < 8; ++r) {
      int row = row0 + r + 8 * grp;
      if (row < n) O[(size_t)row * 128 + n0 + ln] = (_Float16)(acc[r] * scale);
    }
  }
}

__global__ __launch_bounds__(128) void k_qkv(
    const float* __restrict__ x, const float* __restrict__ Wq,
    const float* __restrict__ Wk, const float* __restrict__ Wv,
    _Float16* __restrict__ qh, _Float16* __restrict__ kh,
    _Float16* __restrict__ vh, int n) {
  __shared__ _Float16 xa[16 * 128];
  int row0 = blockIdx.x * 16, t = threadIdx.x;
  for (int i = t; i < 16 * 128; i += 128) {
    int r = i >> 7, c = i & 127;
    int rr = min(row0 + r, n - 1);
    xa[i] = (_Float16)x[(size_t)rr * 128 + c];
  }
  __syncthreads();
  int wave = t >> 5, lane = t & 31, grp = lane >> 4, ln = lane & 15;
  qkv_one(xa, Wq, qh, row0, wave, grp, ln, 0.25f, n);  // fold 1/sqrt(D)=0.25
  qkv_one(xa, Wk, kh, row0, wave, grp, ln, 1.0f, n);
  qkv_one(xa, Wv, vh, row0, wave, grp, ln, 1.0f, n);
}

// ---------------------------------------------------------------------------
// Kernel 2: init softmax state + attention accumulator
// ---------------------------------------------------------------------------
__global__ void k_init(unsigned* __restrict__ smax, float* __restrict__ ssum,
                       float* __restrict__ attn, int nsm, int nat) {
  int i = blockIdx.x * blockDim.x + threadIdx.x;
  if (i < nsm) {
    smax[i] = 0u;  // encoded -inf
    ssum[i] = 0.0f;
  }
  if (i < nat) attn[i] = 0.0f;
}

// ---------------------------------------------------------------------------
// Kernel 3: edge scores + segment max. thread per (edge, head).
// edges e<nE from edge_index; e>=nE are self loops (row=col=e-nE).
// ---------------------------------------------------------------------------
__global__ void k_score(const int* __restrict__ ei,
                        const _Float16* __restrict__ qh,
                        const _Float16* __restrict__ kh,
                        float* __restrict__ scores, unsigned* __restrict__ smax,
                        int nE, int nT) {
  long long id = (long long)blockIdx.x * blockDim.x + threadIdx.x;
  if (id >= (long long)nT * 8) return;
  int e = (int)(id >> 3), h = (int)(id & 7);
  int row, col;
  if (e < nE) { row = ei[e]; col = ei[nE + e]; }
  else        { row = col = e - nE; }
  const v8h* qp = (const v8h*)(qh + (size_t)row * 128 + h * 16);
  const v8h* kp = (const v8h*)(kh + (size_t)col * 128 + h * 16);
  v8h q0 = qp[0], q1 = qp[1], k0 = kp[0], k1 = kp[1];
  float s = 0.0f;
#pragma unroll
  for (int j = 0; j < 8; ++j)
    s += (float)q0[j] * (float)k0[j] + (float)q1[j] * (float)k1[j];
  scores[(size_t)e * 8 + h] = s;
  atomicMax(&smax[col * 8 + h], fenc(s));
}

// ---------------------------------------------------------------------------
// Kernel 4: exp(score - max) + segment sum (in-place over scores buffer)
// ---------------------------------------------------------------------------
__global__ void k_expsum(const int* __restrict__ ei,
                         float* __restrict__ scores,
                         const unsigned* __restrict__ smax,
                         float* __restrict__ ssum, int nE, int nT) {
  long long id = (long long)blockIdx.x * blockDim.x + threadIdx.x;
  if (id >= (long long)nT * 8) return;
  int e = (int)(id >> 3), h = (int)(id & 7);
  int col = (e < nE) ? ei[nE + e] : (e - nE);
  float m = fdec(smax[col * 8 + h]);
  float w = __expf(scores[(size_t)e * 8 + h] - m);
  scores[(size_t)e * 8 + h] = w;
  atomicAdd(&ssum[col * 8 + h], w);
}

// ---------------------------------------------------------------------------
// Kernel 5: attn[col] += alpha * v[row]. thread per (edge, channel).
// ---------------------------------------------------------------------------
__global__ void k_accum(const int* __restrict__ ei,
                        const _Float16* __restrict__ vh,
                        const float* __restrict__ expw,
                        const float* __restrict__ ssum,
                        float* __restrict__ attn, int nE, int nT) {
  long long id = (long long)blockIdx.x * blockDim.x + threadIdx.x;
  if (id >= (long long)nT * 128) return;
  int e = (int)(id >> 7), c = (int)(id & 127), h = c >> 4;
  int row, col;
  if (e < nE) { row = ei[e]; col = ei[nE + e]; }
  else        { row = col = e - nE; }
  float alpha = expw[(size_t)e * 8 + h] / (ssum[col * 8 + h] + 1e-8f);
  atomicAdd(&attn[(size_t)col * 128 + c],
            alpha * (float)vh[(size_t)row * 128 + c]);
}

// ---------------------------------------------------------------------------
// Kernel 6: x1 = LayerNorm(x + attn @ Wo + bo) — fused GEMM + residual + LN
// ---------------------------------------------------------------------------
__global__ __launch_bounds__(128) void k_proj_ln(
    const float* __restrict__ attn, const float* __restrict__ Wo,
    const float* __restrict__ bo, const float* __restrict__ x,
    const float* __restrict__ g1, const float* __restrict__ b1,
    float* __restrict__ x1, int n) {
  __shared__ _Float16 xa[16 * 128];
  __shared__ float pre[16 * 128];
  __shared__ float mu[16], rs[16];
  int row0 = blockIdx.x * 16, t = threadIdx.x;
  for (int i = t; i < 16 * 128; i += 128) {
    int r = i >> 7, c = i & 127;
    int rr = min(row0 + r, n - 1);
    xa[i] = (_Float16)attn[(size_t)rr * 128 + c];
  }
  __syncthreads();
  int wave = t >> 5, lane = t & 31, grp = lane >> 4, ln = lane & 15;
#pragma unroll
  for (int nt = 0; nt < 2; ++nt) {
    int n0 = (wave * 2 + nt) * 16;
    v8f acc = {};
#pragma unroll
    for (int ks = 0; ks < 4; ++ks) {
      v16h a = load_a_lds(xa, ln, ks * 32, 128, grp);
      v16h b = load_b_gl(Wo, ks * 32, n0 + ln, 128, grp);
      acc = wmma16(a, b, acc);
    }
    float bb = bo[n0 + ln];
#pragma unroll
    for (int r = 0; r < 8; ++r) pre[(r + 8 * grp) * 128 + n0 + ln] = acc[r] + bb;
  }
  __syncthreads();
  for (int i = t; i < 16 * 128; i += 128) {
    int r = i >> 7, c = i & 127;
    int rr = min(row0 + r, n - 1);
    pre[i] += x[(size_t)rr * 128 + c];
  }
  __syncthreads();
  if (t < 16) {
    float s = 0.0f, ss = 0.0f;
    for (int c = 0; c < 128; ++c) {
      float v = pre[t * 128 + c];
      s += v; ss += v * v;
    }
    float m = s * (1.0f / 128.0f);
    float var = ss * (1.0f / 128.0f) - m * m;
    mu[t] = m;
    rs[t] = rsqrtf(var + 1e-5f);
  }
  __syncthreads();
  for (int i = t; i < 16 * 128; i += 128) {
    int r = i >> 7, c = i & 127;
    int row = row0 + r;
    if (row < n)
      x1[(size_t)row * 128 + c] = (pre[i] - mu[r]) * rs[r] * g1[c] + b1[c];
  }
}

// ---------------------------------------------------------------------------
// Kernel 7: fused FFN: out = LN2(x1 + gelu(x1@Wf1+bf1)@Wf2 + bf2)
// 16x512 hidden kept in LDS as f16; both GEMMs via WMMA.
// ---------------------------------------------------------------------------
__global__ __launch_bounds__(128) void k_ffn(
    const float* __restrict__ x1, const float* __restrict__ Wf1,
    const float* __restrict__ bf1, const float* __restrict__ Wf2,
    const float* __restrict__ bf2, const float* __restrict__ g2,
    const float* __restrict__ b2, float* __restrict__ out, int n) {
  __shared__ _Float16 xa[16 * 128];   // 4 KB
  __shared__ _Float16 hh[16 * 512];   // 16 KB hidden (f16)
  __shared__ float pre[16 * 128];     // 8 KB
  __shared__ float mu[16], rs[16];
  int row0 = blockIdx.x * 16, t = threadIdx.x;
  for (int i = t; i < 16 * 128; i += 128) {
    int r = i >> 7, c = i & 127;
    int rr = min(row0 + r, n - 1);
    xa[i] = (_Float16)x1[(size_t)rr * 128 + c];
  }
  __syncthreads();
  int wave = t >> 5, lane = t & 31, grp = lane >> 4, ln = lane & 15;
  // GEMM1: hidden = gelu(x1 @ Wf1 + bf1); 32 col-tiles, 8 per wave
#pragma unroll
  for (int nt = 0; nt < 8; ++nt) {
    int n0 = (wave * 8 + nt) * 16;
    v8f acc = {};
#pragma unroll
    for (int ks = 0; ks < 4; ++ks) {
      v16h a = load_a_lds(xa, ln, ks * 32, 128, grp);
      v16h b = load_b_gl(Wf1, ks * 32, n0 + ln, 512, grp);
      acc = wmma16(a, b, acc);
    }
    float bb = bf1[n0 + ln];
#pragma unroll
    for (int r = 0; r < 8; ++r) {
      float v = acc[r] + bb;
      v = 0.5f * v * (1.0f + erff(v * 0.70710678118654752f));  // exact gelu
      hh[(r + 8 * grp) * 512 + n0 + ln] = (_Float16)v;
    }
  }
  __syncthreads();
  // GEMM2: pre = hidden @ Wf2 + bf2; K=512 (16 k-steps), 8 col-tiles, 2/wave
#pragma unroll
  for (int nt = 0; nt < 2; ++nt) {
    int n0 = (wave * 2 + nt) * 16;
    v8f acc = {};
    for (int ks = 0; ks < 16; ++ks) {
      v16h a = load_a_lds(hh, ln, ks * 32, 512, grp);
      v16h b = load_b_gl(Wf2, ks * 32, n0 + ln, 128, grp);
      acc = wmma16(a, b, acc);
    }
    float bb = bf2[n0 + ln];
#pragma unroll
    for (int r = 0; r < 8; ++r) pre[(r + 8 * grp) * 128 + n0 + ln] = acc[r] + bb;
  }
  __syncthreads();
  // residual add (reload x1) + LN2
  for (int i = t; i < 16 * 128; i += 128) {
    int r = i >> 7, c = i & 127;
    int rr = min(row0 + r, n - 1);
    pre[i] += x1[(size_t)rr * 128 + c];
  }
  __syncthreads();
  if (t < 16) {
    float s = 0.0f, ss = 0.0f;
    for (int c = 0; c < 128; ++c) {
      float v = pre[t * 128 + c];
      s += v; ss += v * v;
    }
    float m = s * (1.0f / 128.0f);
    float var = ss * (1.0f / 128.0f) - m * m;
    mu[t] = m;
    rs[t] = rsqrtf(var + 1e-5f);
  }
  __syncthreads();
  for (int i = t; i < 16 * 128; i += 128) {
    int r = i >> 7, c = i & 127;
    int row = row0 + r;
    if (row < n)
      out[(size_t)row * 128 + c] = (pre[i] - mu[r]) * rs[r] * g2[c] + b2[c];
  }
}

// ---------------------------------------------------------------------------
// Host launcher
// ---------------------------------------------------------------------------
extern "C" void kernel_launch(void* const* d_in, const int* in_sizes, int n_in,
                              void* d_out, int out_size, void* d_ws,
                              size_t ws_size, hipStream_t stream) {
  const float* x   = (const float*)d_in[0];
  const int*   ei  = (const int*)d_in[1];   // (2,E) int32 (jax x64 disabled)
  const float* Wq  = (const float*)d_in[2];
  const float* Wk  = (const float*)d_in[3];
  const float* Wv  = (const float*)d_in[4];
  const float* Wo  = (const float*)d_in[5];
  const float* bo  = (const float*)d_in[6];
  const float* Wf1 = (const float*)d_in[7];
  const float* bf1 = (const float*)d_in[8];
  const float* Wf2 = (const float*)d_in[9];
  const float* bf2 = (const float*)d_in[10];
  const float* g1  = (const float*)d_in[11];
  const float* b1  = (const float*)d_in[12];
  const float* g2  = (const float*)d_in[13];
  const float* b2  = (const float*)d_in[14];

  const int n  = in_sizes[0] / 128;   // 50000
  const int nE = in_sizes[1] / 2;     // 800000
  const int nT = nE + n;              // edges + self loops

  char* w = (char*)d_ws;
  size_t off = 0;
  auto alloc = [&](size_t bytes) -> char* {
    char* p = w + off;
    off = (off + bytes + 255) & ~(size_t)255;
    return p;
  };
  _Float16* qh   = (_Float16*)alloc((size_t)n * 128 * sizeof(_Float16));
  _Float16* kh   = (_Float16*)alloc((size_t)n * 128 * sizeof(_Float16));
  _Float16* vh   = (_Float16*)alloc((size_t)n * 128 * sizeof(_Float16));
  unsigned* smax = (unsigned*)alloc((size_t)n * 8 * sizeof(unsigned));
  float*    ssum = (float*)   alloc((size_t)n * 8 * sizeof(float));
  float*    expw = (float*)   alloc((size_t)nT * 8 * sizeof(float));
  float*    attn = (float*)   alloc((size_t)n * 128 * sizeof(float));
  float*    x1   = (float*)   alloc((size_t)n * 128 * sizeof(float));
  float*    outp = (float*)d_out;

  const int tiles = (n + 15) / 16;

  k_qkv<<<tiles, 128, 0, stream>>>(x, Wq, Wk, Wv, qh, kh, vh, n);

  {
    int tot = n * 128;
    k_init<<<(tot + 255) / 256, 256, 0, stream>>>(smax, ssum, attn, n * 8, tot);
  }
  {
    long long tot = (long long)nT * 8;
    int blocks = (int)((tot + 255) / 256);
    k_score<<<blocks, 256, 0, stream>>>(ei, qh, kh, expw, smax, nE, nT);
    k_expsum<<<blocks, 256, 0, stream>>>(ei, expw, smax, ssum, nE, nT);
  }
  {
    long long tot = (long long)nT * 128;
    int blocks = (int)((tot + 255) / 256);
    k_accum<<<blocks, 256, 0, stream>>>(ei, vh, expw, ssum, attn, nE, nT);
  }

  k_proj_ln<<<tiles, 128, 0, stream>>>(attn, Wo, bo, x, g1, b1, x1, n);
  k_ffn<<<tiles, 128, 0, stream>>>(x1, Wf1, bf1, Wf2, bf2, g2, b2, outp, n);
}
